// CapsuleNetwork_24206435680414
// MI455X (gfx1250) — matrix operand order repaired and checked by reading, not compiled, not run on hardware
//
#include <hip/hip_runtime.h>
#include <hip/hip_bf16.h>
#include <math.h>

typedef __attribute__((ext_vector_type(16))) _Float16 v16h;
typedef __attribute__((ext_vector_type(8)))  _Float16 v8h;
typedef __attribute__((ext_vector_type(8)))  float    v8f;

#define EPSF 1e-7f

// ---------------------------------------------------------------------------
// Kernel 1: conv1 9x9 VALID stride1, 1->256 channels, + bias + ReLU, f16 out.
// x: (128,32,32,1) f32   w: (9,9,1,256) f32   out: (128,24,24,256) f16
// One block = 4x4 output-pixel tile x 256 channels. Input patch (12x12) in LDS.
// ---------------------------------------------------------------------------
__global__ __launch_bounds__(256) void conv1_kernel(
    const float* __restrict__ x, const float* __restrict__ w,
    const float* __restrict__ bias, _Float16* __restrict__ out)
{
    __shared__ float lin[144];                // 12x12 input patch
    int t  = blockIdx.x;                      // 128*36 blocks
    int b  = t / 36, tr = t % 36;
    int oh0 = (tr / 6) * 4, ow0 = (tr % 6) * 4;
    int tid = threadIdx.x;
    if (tid < 144) {
        int r = tid / 12, cc = tid % 12;
        lin[tid] = x[(b * 32 + oh0 + r) * 32 + ow0 + cc];
    }
    __syncthreads();
    int c = tid;                              // output channel
    float bv = bias[c];
    float acc[16];
#pragma unroll
    for (int p = 0; p < 16; ++p) acc[p] = bv;
    for (int i = 0; i < 81; ++i) {
        int kh = i / 9, kw = i % 9;
        float wv = w[i * 256 + c];            // coalesced across threads
#pragma unroll
        for (int p = 0; p < 16; ++p) {
            int pr = p >> 2, pc = p & 3;
            acc[p] += lin[(pr + kh) * 12 + (pc + kw)] * wv;
        }
    }
#pragma unroll
    for (int p = 0; p < 16; ++p) {
        int pr = p >> 2, pc = p & 3;
        float v = acc[p] > 0.f ? acc[p] : 0.f;
        out[(((size_t)b * 24 + oh0 + pr) * 24 + (ow0 + pc)) * 256 + c] = (_Float16)v;
    }
}

// ---------------------------------------------------------------------------
// Kernel 2: repack conv2_w (9,9,256,1024) f32 -> f16 in WMMA B-fragment layout.
// Bpack[kk][nt][lane][j] = W[kk*32 + (lane>=16?16:0) + j][nt*16 + (lane&15)]
// kk in [0,648) (K/32), nt in [0,64) (N/16). Per-lane 16 contiguous halfs.
// ---------------------------------------------------------------------------
__global__ __launch_bounds__(256) void packw_kernel(
    const float* __restrict__ w, _Float16* __restrict__ bp)
{
    size_t tid = (size_t)blockIdx.x * 256 + threadIdx.x;  // 648*64*32*16 total
    int j    = (int)(tid & 15);
    int lane = (int)((tid >> 4) & 31);
    int nt   = (int)((tid >> 9) & 63);
    int kk   = (int)(tid >> 15);
    int kg   = kk * 32 + ((lane >> 4) << 4) + j;
    int n    = nt * 16 + (lane & 15);
    bp[tid]  = (_Float16)w[(size_t)kg * 1024 + n];
}

// ---------------------------------------------------------------------------
// Kernel 3: conv2 as implicit GEMM on v_wmma_f32_16x16x32_f16.
// M=18432 (b,oh,ow), N=1024 (co), K=20736 (kh,kw,ci). stride 2, SAME pad(3,4).
// One wave owns a 64x64 tile -> 16 v8f accumulators; 648 K-steps of 32.
// ---------------------------------------------------------------------------
__global__ __launch_bounds__(128) void conv2_gemm_kernel(
    const _Float16* __restrict__ A,   // conv1 f16 out (128,24,24,256)
    const _Float16* __restrict__ Bp,  // packed weights
    const float* __restrict__ bias,   // conv2_b (1024)
    float* __restrict__ U)            // (18432,1024) f32
{
    int wave = threadIdx.x >> 5;
    int lane = threadIdx.x & 31;
    int tile = blockIdx.x * 4 + wave;     // 4608 tiles = 288 mtiles x 16 ntiles
    int mt   = tile >> 4;
    int nt4  = (tile & 15) * 4;           // first 16-wide N-fragment index
    int mbase = mt * 64;
    int laneoff = (lane < 16) ? 0 : 16;   // A layout K sub-chunk select
    int lm = lane & 15;

    int pixbase[4], ihb[4], iwb[4];
#pragma unroll
    for (int i = 0; i < 4; ++i) {
        int m = mbase + i * 16 + lm;      // A row for this lane
        int b = m / 144, r = m % 144;
        int oh = r / 12, ow = r % 12;
        ihb[i] = oh * 2 - 3;
        iwb[i] = ow * 2 - 3;
        pixbase[i] = b * 576;             // b*24*24
    }

    v8f zf = {};
    v8f acc[16];
#pragma unroll
    for (int i = 0; i < 16; ++i) acc[i] = zf;

    for (int khkw = 0; khkw < 81; ++khkw) {
        int kh = khkw / 9, kw = khkw % 9;
        int abase[4]; bool aval[4];
#pragma unroll
        for (int i = 0; i < 4; ++i) {
            int ih = ihb[i] + kh, iw = iwb[i] + kw;
            aval[i]  = ((unsigned)ih < 24u) && ((unsigned)iw < 24u);
            abase[i] = ((pixbase[i] + ih * 24 + iw) << 8) + laneoff;
        }
        const _Float16* bpb = Bp + ((size_t)(khkw * 8) * 64 + nt4) * 512 + lane * 16;
        for (int cs = 0; cs < 8; ++cs) {
            int ci = cs * 32;
            union Frag { v16h v; struct { v8h lo, hi; } p; } Af[4], Bf[4];
            v8h zh = {};
#pragma unroll
            for (int i = 0; i < 4; ++i) {
                v8h lo = zh, hi = zh;
                if (aval[i]) {                    // zero-pad SAME border
                    const _Float16* ap = A + abase[i] + ci;
                    lo = *(const v8h*)(ap);       // K +0..7   (this lane group)
                    hi = *(const v8h*)(ap + 8);   // K +8..15  (this lane group)
                }
                Af[i].p.lo = lo; Af[i].p.hi = hi;
            }
            const _Float16* bq = bpb + (size_t)cs * 64 * 512;
#pragma unroll
            for (int j = 0; j < 4; ++j) {
                Bf[j].p.lo = *(const v8h*)(bq + j * 512);
                Bf[j].p.hi = *(const v8h*)(bq + j * 512 + 8);
            }
#pragma unroll
            for (int i = 0; i < 4; ++i)
#pragma unroll
                for (int j = 0; j < 4; ++j)
                    acc[i * 4 + j] = __builtin_amdgcn_wmma_f32_16x16x32_f16(
                        false, Af[i].v, false, Bf[j].v,
                        (short)0, acc[i * 4 + j], false, false);
        }
    }

    // Epilogue: C layout -> lane<16: N=lane, M=r; lane>=16: N=lane-16, M=8+r.
#pragma unroll
    for (int j = 0; j < 4; ++j) {
        int n = (nt4 + j) * 16 + lm;
        float bv = bias[n];
#pragma unroll
        for (int i = 0; i < 4; ++i) {
            int mrow = mbase + i * 16 + ((lane < 16) ? 0 : 8);
            v8f a = acc[i * 4 + j];
#pragma unroll
            for (int r = 0; r < 8; ++r)
                U[(size_t)(mrow + r) * 1024 + n] = a[r] + bv;
        }
    }
}

// ---------------------------------------------------------------------------
// Kernel 4: u_hat[b,n,kd] = sum_p w[n,kd,p] * u[b,n,p]   (kd = k*16+d, p<32)
// Block = (b, 8 capsules); u rows staged in LDS; w reads coalesced b128.
// ---------------------------------------------------------------------------
__global__ __launch_bounds__(256) void uhat_kernel(
    const float* __restrict__ u, const float* __restrict__ w,
    float* __restrict__ uhat)
{
    __shared__ float us[8][32];
    int bid = blockIdx.x;                 // 128*576
    int b = bid / 576, g = bid % 576;
    int tid = threadIdx.x;
    us[tid >> 5][tid & 31] = u[(size_t)b * 147456 + g * 256 + tid];
    __syncthreads();
    int nl = tid >> 5, kd = tid & 31;
    int n = g * 8 + nl;
    const float* wp = w + (size_t)n * 1024 + kd * 32;
    float acc = 0.f;
#pragma unroll
    for (int p = 0; p < 32; ++p) acc += wp[p] * us[nl][p];
    uhat[(size_t)b * 147456 + n * 32 + kd] = acc;
}

// ---------------------------------------------------------------------------
// Kernel 5: dynamic routing, 3 iterations. One block per batch row.
// b_logits (4608x2) live in LDS the whole time. Writes v -> d_out[0:4096).
// ---------------------------------------------------------------------------
__global__ __launch_bounds__(256) void routing_kernel(
    const float* __restrict__ uhat, float* __restrict__ vout)
{
    __shared__ float bl[4608 * 2];
    __shared__ float spart[8][32];
    __shared__ float vs[32];
    __shared__ float vv[32];
    int b = blockIdx.x;
    int tid = threadIdx.x;
    int lane = tid & 31, wv = tid >> 5;
    const float* ub = uhat + (size_t)b * 147456;
    for (int n = tid; n < 9216; n += 256) bl[n] = 0.f;
    __syncthreads();
    for (int it = 0; it < 3; ++it) {
        float s[32];
#pragma unroll
        for (int i = 0; i < 32; ++i) s[i] = 0.f;
        for (int n = tid; n < 4608; n += 256) {
            float b0 = bl[n * 2], b1 = bl[n * 2 + 1];
            float mx = fmaxf(b0, b1);
            float e0 = expf(b0 - mx), e1 = expf(b1 - mx);
            float inv = 1.f / (e0 + e1);
            float c0 = e0 * inv, c1 = e1 * inv;
            const float* uh = ub + n * 32;
#pragma unroll
            for (int d = 0; d < 16; ++d) {
                s[d]      += c0 * uh[d];
                s[16 + d] += c1 * uh[16 + d];
            }
        }
#pragma unroll
        for (int off = 16; off >= 1; off >>= 1)
#pragma unroll
            for (int i = 0; i < 32; ++i) s[i] += __shfl_xor(s[i], off, 32);
        if (lane == 0)
            for (int i = 0; i < 32; ++i) spart[wv][i] = s[i];
        __syncthreads();
        if (tid < 32) {
            float ss = 0.f;
            for (int w8 = 0; w8 < 8; ++w8) ss += spart[w8][tid];
            vs[tid] = ss;
        }
        __syncthreads();
        if (tid < 32) {                   // squash per k
            int k = tid >> 4;
            float nn = 0.f;
            for (int d = 0; d < 16; ++d) { float t = vs[k * 16 + d]; nn += t * t; }
            float nrm = sqrtf(nn);
            float scale = (nn / (1.f + nn)) / (nrm + EPSF);
            vv[tid] = vs[tid] * scale;
        }
        __syncthreads();
        float vr[32];
#pragma unroll
        for (int i = 0; i < 32; ++i) vr[i] = vv[i];
        for (int n = tid; n < 4608; n += 256) {
            const float* uh = ub + n * 32;
            float a0 = 0.f, a1 = 0.f;
#pragma unroll
            for (int d = 0; d < 16; ++d) {
                a0 += uh[d] * vr[d];
                a1 += uh[16 + d] * vr[16 + d];
            }
            bl[n * 2]     += a0;
            bl[n * 2 + 1] += a1;
        }
        __syncthreads();
    }
    if (tid < 32) vout[b * 32 + tid] = vv[tid];
}

// ---------------------------------------------------------------------------
// Decoder (tiny): h1 = relu(mask(v) @ d1_w + b1); h2 = relu(h1 @ d2_w + b2);
// recon = sigmoid(h2 @ d3_w + b3) -> d_out[4096:].
// ---------------------------------------------------------------------------
__global__ __launch_bounds__(256) void dec1_kernel(
    const float* __restrict__ vout, const float* __restrict__ y,
    const float* __restrict__ w, const float* __restrict__ bias,
    float* __restrict__ h1)
{
    int b = blockIdx.x, j = threadIdx.x;
    float y0 = y[b * 2], y1 = y[b * 2 + 1];
    float acc = bias[j];
    const float* vb = vout + b * 32;
#pragma unroll
    for (int i = 0; i < 32; ++i)
        acc += vb[i] * (i < 16 ? y0 : y1) * w[i * 256 + j];
    h1[b * 256 + j] = fmaxf(acc, 0.f);
}

__global__ __launch_bounds__(512) void dec2_kernel(
    const float* __restrict__ h1, const float* __restrict__ w,
    const float* __restrict__ bias, float* __restrict__ h2)
{
    int b = blockIdx.x, j = threadIdx.x;
    float acc = bias[j];
    const float* hb = h1 + b * 256;
    for (int i = 0; i < 256; ++i) acc += hb[i] * w[i * 512 + j];
    h2[b * 512 + j] = fmaxf(acc, 0.f);
}

__global__ __launch_bounds__(1024) void dec3_kernel(
    const float* __restrict__ h2, const float* __restrict__ w,
    const float* __restrict__ bias, float* __restrict__ recon)
{
    int b = blockIdx.x, j = threadIdx.x;
    float acc = bias[j];
    const float* hb = h2 + b * 512;
    for (int i = 0; i < 512; ++i) acc += hb[i] * w[i * 1024 + j];
    recon[b * 1024 + j] = 1.f / (1.f + expf(-acc));
}

// ---------------------------------------------------------------------------
extern "C" void kernel_launch(void* const* d_in, const int* in_sizes, int n_in,
                              void* d_out, int out_size, void* d_ws, size_t ws_size,
                              hipStream_t stream)
{
    const float* x     = (const float*)d_in[0];
    const float* y     = (const float*)d_in[1];
    const float* c1w   = (const float*)d_in[2];
    const float* c1b   = (const float*)d_in[3];
    const float* c2w   = (const float*)d_in[4];
    const float* c2b   = (const float*)d_in[5];
    const float* wcaps = (const float*)d_in[6];
    const float* d1w   = (const float*)d_in[7];
    const float* d1b   = (const float*)d_in[8];
    const float* d2w   = (const float*)d_in[9];
    const float* d2b   = (const float*)d_in[10];
    const float* d3w   = (const float*)d_in[11];
    const float* d3b   = (const float*)d_in[12];
    float* out = (float*)d_out;

    // Workspace layout (bytes), all 256B-aligned, total ~221 MB:
    char* ws = (char*)d_ws;
    _Float16* conv1f = (_Float16*)(ws);                 //  37,748,736 B
    _Float16* bpack  = (_Float16*)(ws + 37748736);      //  42,467,328 B
    float* u    = (float*)(ws + 80216064);              //  75,497,472 B
    float* uhat = (float*)(ws + 155713536);             //  75,497,472 B
    float* h1   = (float*)(ws + 231211008);             //     131,072 B
    float* h2   = (float*)(ws + 231342080);             //     262,144 B

    conv1_kernel     <<<4608,  256, 0, stream>>>(x, c1w, c1b, conv1f);
    packw_kernel     <<<82944, 256, 0, stream>>>(c2w, bpack);
    conv2_gemm_kernel<<<1152,  128, 0, stream>>>(conv1f, bpack, c2b, u);
    uhat_kernel      <<<73728, 256, 0, stream>>>(u, wcaps, uhat);
    routing_kernel   <<<128,   256, 0, stream>>>(uhat, out);
    dec1_kernel      <<<128,   256, 0, stream>>>(out, y, d1w, d1b, h1);
    dec2_kernel      <<<128,   512, 0, stream>>>(h1, d2w, d2b, h2);
    dec3_kernel      <<<128,  1024, 0, stream>>>(h2, d3w, d3b, out + 4096);
}
